// CSSM_81595788689879
// MI455X (gfx1250) — matrix operand order repaired and checked by reading, not compiled, not run on hardware
//
#include <hip/hip_runtime.h>
#include <hip/hip_bf16.h>
#include <utility>

// ---------------------------------------------------------------------------
// CDNA5 (gfx1250) implementation of the CSSM multimodal classifier forward.
// Dense linears run through v_wmma_f32_16x16x32_f16 (wave32 WMMA),
// 32x32 tile per wave (2x2 wmma sub-tiles), vectorized b128 global loads
// and b128/b64 LDS staging on the full-tile fast path.
// ---------------------------------------------------------------------------

#define NTOK 169
#define BSZ  256

typedef __attribute__((ext_vector_type(16))) _Float16 v16h;
typedef __attribute__((ext_vector_type(8)))  _Float16 v8h;
typedef __attribute__((ext_vector_type(4)))  _Float16 v4h;
typedef __attribute__((ext_vector_type(8)))  float    v8f;
typedef __attribute__((ext_vector_type(4)))  float    v4f;

enum { ACT_NONE = 0, ACT_RELU = 1, ACT_SIGMOID = 2, ACT_GELU = 3 };

// ---------------------------------------------------------------------------
// Generic WMMA GEMM:  Y[m,f] = epi( sum_k X[m,k] * W[f,k] )
//   epi: +bias[f] -> activation -> +res[m,f] -> +pos[m%NTOK, f]
// One wave (32 threads) computes a 32x32 tile (2x2 wmma accumulators),
// K loop in chunks of 32 staged through LDS as f16.
// ---------------------------------------------------------------------------
__global__ __launch_bounds__(32) void gemm_wmma_kernel(
    const float* __restrict__ X, int ldx,
    const float* __restrict__ W,              // [Ft, Kt] row-major
    const float* __restrict__ bias,           // [Ft] or null
    const float* __restrict__ res, int ldres, // [Mt, ldres] or null
    const float* __restrict__ pos,            // [NTOK, Ft] or null
    float* __restrict__ Y, int ldy,
    int Mt, int Kt, int Ft, int act)
{
    __shared__ _Float16 As[32][40];   // [m][k]  (row-major, +pad)
    __shared__ _Float16 Bs[32][40];   // [k][f]  (transposed,  +pad)

    const int tid = threadIdx.x;
    const int m0  = blockIdx.y * 32;
    const int f0  = blockIdx.x * 32;
    const int hlf = tid >> 4;         // half-wave select
    const int lm  = tid & 15;

    const bool full = (m0 + 32 <= Mt) && (f0 + 32 <= Ft);

    v8f acc[2][2];
    acc[0][0] = (v8f){}; acc[0][1] = (v8f){};
    acc[1][0] = (v8f){}; acc[1][1] = (v8f){};

    for (int k0 = 0; k0 < Kt; k0 += 32) {
        if (full && (k0 + 32 <= Kt)) {
            // ---------------- fast path: fully in-bounds tile ----------
            // A tile: lane = row, 32 consecutive k as 8x b128 loads,
            //         packed f16 stores as 4x b128.
            {
                const float* xrow = X + (size_t)(m0 + tid) * ldx + k0;
                if (k0 + 32 < Kt)
                    __builtin_prefetch(xrow + 32, 0, 3);
                #pragma unroll
                for (int t = 0; t < 4; ++t) {
                    v4f lo = *(const v4f*)(xrow + t * 8);
                    v4f hi = *(const v4f*)(xrow + t * 8 + 4);
                    v8h h;
                    #pragma unroll
                    for (int j = 0; j < 4; ++j) {
                        h[j]     = (_Float16)lo[j];
                        h[4 + j] = (_Float16)hi[j];
                    }
                    *(v8h*)&As[tid][t * 8] = h;
                }
            }
            // B tile: lane covers a 4(f) x 8(k) sub-block; transpose in
            // registers -> ds_store_b64 per k-column.
            {
                const int fb = (tid & 7) * 4;
                const int kb = (tid >> 3) * 8;
                float wv[4][8];
                #pragma unroll
                for (int i = 0; i < 4; ++i) {
                    const float* wrow = W + (size_t)(f0 + fb + i) * Kt + k0 + kb;
                    if (k0 + 32 < Kt)
                        __builtin_prefetch(wrow + 32, 0, 3);
                    v4f lo = *(const v4f*)(wrow);
                    v4f hi = *(const v4f*)(wrow + 4);
                    #pragma unroll
                    for (int j = 0; j < 4; ++j) {
                        wv[i][j]     = lo[j];
                        wv[i][4 + j] = hi[j];
                    }
                }
                #pragma unroll
                for (int kk = 0; kk < 8; ++kk) {
                    v4h h;
                    #pragma unroll
                    for (int i = 0; i < 4; ++i) h[i] = (_Float16)wv[i][kk];
                    *(v4h*)&Bs[kb + kk][fb] = h;
                }
            }
        } else {
            // ---------------- guarded edge path ------------------------
            const int kk = tid;
            #pragma unroll 4
            for (int r = 0; r < 32; ++r) {
                const int m = m0 + r, k = k0 + kk;
                As[r][kk] = (m < Mt && k < Kt)
                          ? (_Float16)X[(size_t)m * ldx + k] : (_Float16)0.f;
                const int f = f0 + r;
                Bs[kk][r] = (f < Ft && k < Kt)
                          ? (_Float16)W[(size_t)f * Kt + k] : (_Float16)0.f;
            }
        }
        __syncthreads();

        // ---- fragments per CDNA5 16-bit WMMA VGPR layouts --------------
        // A: lane = M row (tile-local), elems stripe K (half-wave split)
        // B: lane = N col, elems stripe K (half-wave split)
        v16h afr[2], bfr[2];
        #pragma unroll
        for (int t = 0; t < 2; ++t) {
            const int row = t * 16 + lm;
            v8h alo = *(const v8h*)&As[row][hlf * 8];
            v8h ahi = *(const v8h*)&As[row][16 + hlf * 8];
            afr[t] = __builtin_shufflevector(alo, ahi,
                     0,1,2,3,4,5,6,7,8,9,10,11,12,13,14,15);
            #pragma unroll
            for (int s = 0; s < 16; ++s)
                bfr[t][s] = Bs[(s < 8 ? hlf * 8 + s : 16 + hlf * 8 + (s - 8))][t * 16 + lm];
        }
        #pragma unroll
        for (int ti = 0; ti < 2; ++ti)
            #pragma unroll
            for (int tj = 0; tj < 2; ++tj)
                acc[ti][tj] = __builtin_amdgcn_wmma_f32_16x16x32_f16(
                    false, afr[ti], false, bfr[tj], (short)0, acc[ti][tj],
                    false, false);
        __syncthreads();
    }

    // ---- epilogue: C/D layout lane = N col, vgpr r -> M = r + 8*half ---
    #pragma unroll
    for (int tj = 0; tj < 2; ++tj) {
        const int col = f0 + tj * 16 + lm;
        if (col >= Ft) continue;
        const float bv = bias ? bias[col] : 0.f;
        const float pv = pos ? 0.f : 0.f;  (void)pv;
        #pragma unroll
        for (int ti = 0; ti < 2; ++ti) {
            #pragma unroll
            for (int r = 0; r < 8; ++r) {
                const int m = m0 + ti * 16 + r + 8 * hlf;
                if (m < Mt) {
                    float v = acc[ti][tj][r] + bv;
                    if      (act == ACT_RELU)    v = fmaxf(v, 0.f);
                    else if (act == ACT_SIGMOID) v = 1.f / (1.f + __expf(-v));
                    else if (act == ACT_GELU)    v = 0.5f * v * (1.f + erff(v * 0.70710678118f));
                    if (res) v += res[(size_t)m * ldres + col];
                    if (pos) v += pos[(size_t)(m % NTOK) * Ft + col];
                    Y[(size_t)m * ldy + col] = v;
                }
            }
        }
    }
}

// ---------------------------------------------------------------------------
// LayerNorm over last dim D (one block per token row)
// ---------------------------------------------------------------------------
__global__ __launch_bounds__(64) void ln_kernel(
    const float* __restrict__ X, const float* __restrict__ g,
    const float* __restrict__ b, float* __restrict__ Y, int D)
{
    const size_t row = blockIdx.x;
    const float* x = X + row * (size_t)D;
    __shared__ float ssum[64], ssq[64];
    const int t = threadIdx.x;
    float s = 0.f, q = 0.f;
    for (int i = t; i < D; i += 64) { float v = x[i]; s += v; q += v * v; }
    ssum[t] = s; ssq[t] = q;
    __syncthreads();
    for (int o = 32; o > 0; o >>= 1) {
        if (t < o) { ssum[t] += ssum[t + o]; ssq[t] += ssq[t + o]; }
        __syncthreads();
    }
    const float mean = ssum[0] / D;
    const float var  = ssq[0] / D - mean * mean;
    const float inv  = rsqrtf(var + 1e-5f);
    for (int i = t; i < D; i += 64)
        Y[row * (size_t)D + i] = (x[i] - mean) * inv * g[i] + b[i];
}

// ---------------------------------------------------------------------------
// Depthwise causal conv (kernel 4, left pad 3) + SiLU.  U has ld=ldu.
// ---------------------------------------------------------------------------
__global__ void conv_silu_kernel(const float* __restrict__ U, int ldu,
                                 const float* __restrict__ cw,
                                 const float* __restrict__ cb,
                                 float* __restrict__ UC, int di, size_t total)
{
    size_t i = (size_t)blockIdx.x * blockDim.x + threadIdx.x;
    if (i >= total) return;
    const int c = (int)(i % di);
    const size_t bn = i / di;
    const int n = (int)(bn % NTOK);
    const size_t b = bn / NTOK;
    float acc = cb[c];
    #pragma unroll
    for (int j = 0; j < 4; ++j) {
        const int nn = n - 3 + j;
        if (nn >= 0) acc += cw[c * 4 + j] * U[((size_t)b * NTOK + nn) * ldu + c];
    }
    UC[i] = acc * (1.f / (1.f + __expf(-acc)));
}

// ---------------------------------------------------------------------------
// Mamba decay scan fused with sigmoid gating.
//   s = s*a + bb[n];  y[n] = s * cc[n] * sigmoid(gate[n])
// ---------------------------------------------------------------------------
__global__ void mamba_scan_kernel(const float* __restrict__ BC, int ldbc,
                                  const float* __restrict__ gate, int ldg,
                                  const float* __restrict__ log_decay,
                                  float* __restrict__ Y, int di)
{
    const int i = blockIdx.x * blockDim.x + threadIdx.x;
    if (i >= BSZ * di) return;
    const int c = i % di;
    const int b = i / di;
    const float ld = log_decay[c];
    const float sp = (ld > 20.f) ? ld : log1pf(__expf(ld));   // softplus
    const float a  = __expf(-sp);
    float s = 0.f;
    size_t row = (size_t)b * NTOK;
    for (int n = 0; n < NTOK; ++n, ++row) {
        const float bt = BC[row * ldbc + c];
        const float ct = BC[row * ldbc + di + c];
        s = s * a + bt;
        const float g = gate[row * ldg + c];
        Y[row * (size_t)di + c] = s * ct * (1.f / (1.f + __expf(-g)));
    }
}

// ---------------------------------------------------------------------------
// Multi-head cross attention core (two-pass softmax).  8 heads, dh = E/8.
// Q/K/V/O are [B, NTOK, E], head h occupies cols [h*dh, (h+1)*dh).
// ---------------------------------------------------------------------------
__global__ __launch_bounds__(192) void mha_kernel(
    const float* __restrict__ Q, const float* __restrict__ K,
    const float* __restrict__ V, float* __restrict__ O, int E, int dh)
{
    const int b = blockIdx.x >> 3;
    const int h = blockIdx.x & 7;
    const int r = threadIdx.x;
    if (r >= NTOK) return;
    const size_t base = ((size_t)b * NTOK) * E + (size_t)h * dh;
    const float* q = Q + base + (size_t)r * E;
    float qr[16];
    for (int d = 0; d < dh; ++d) qr[d] = q[d];
    const float scale = rsqrtf((float)dh);
    const float* Kb = K + base;
    const float* Vb = V + base;

    float mx = -1e30f;
    for (int n = 0; n < NTOK; ++n) {
        float s = 0.f;
        for (int d = 0; d < dh; ++d) s += qr[d] * Kb[(size_t)n * E + d];
        mx = fmaxf(mx, s * scale);
    }
    float acc[16];
    for (int d = 0; d < dh; ++d) acc[d] = 0.f;
    float sum = 0.f;
    for (int n = 0; n < NTOK; ++n) {
        float s = 0.f;
        for (int d = 0; d < dh; ++d) s += qr[d] * Kb[(size_t)n * E + d];
        const float w = __expf(s * scale - mx);
        sum += w;
        for (int d = 0; d < dh; ++d) acc[d] += w * Vb[(size_t)n * E + d];
    }
    const float inv = 1.f / sum;
    float* o = O + base + (size_t)r * E;
    for (int d = 0; d < dh; ++d) o[d] = acc[d] * inv;
}

// ---------------------------------------------------------------------------
// Small elementwise / reshape kernels
// ---------------------------------------------------------------------------
__global__ void add_kernel(float* __restrict__ Out, const float* __restrict__ A,
                           const float* __restrict__ Bv, size_t total)
{
    size_t i = (size_t)blockIdx.x * blockDim.x + threadIdx.x;
    if (i < total) Out[i] = A[i] + Bv[i];
}

__global__ void blend_kernel(float* __restrict__ A, const float* __restrict__ O,
                             const float* __restrict__ Mv,
                             const float* __restrict__ alpha, size_t total)
{
    size_t i = (size_t)blockIdx.x * blockDim.x + threadIdx.x;
    if (i >= total) return;
    const float w = 1.f / (1.f + __expf(-alpha[0]));
    A[i] += w * (O[i] + Mv[i]);
}

// X [B, C, NTOK] (channel-major) -> Y [B*NTOK, C], optionally scaled by gate[b,c]
__global__ void tok_transpose_kernel(const float* __restrict__ X,
                                     const float* __restrict__ gate,
                                     float* __restrict__ Y, int C, size_t total)
{
    size_t i = (size_t)blockIdx.x * blockDim.x + threadIdx.x;
    if (i >= total) return;
    const int c = (int)(i % C);
    const size_t bn = i / C;
    const int n = (int)(bn % NTOK);
    const int b = (int)(bn / NTOK);
    float v = X[((size_t)b * C + c) * NTOK + n];
    if (gate) v *= gate[(size_t)b * C + c];
    Y[i] = v;
}

// mean over trailing NTOK of [B, C, NTOK]
__global__ void pool_hw_kernel(const float* __restrict__ X, float* __restrict__ Y,
                               int C)
{
    int i = blockIdx.x * blockDim.x + threadIdx.x;
    if (i >= BSZ * C) return;
    const float* p = X + (size_t)i * NTOK;
    float s = 0.f;
    for (int n = 0; n < NTOK; ++n) s += p[n];
    Y[i] = s / (float)NTOK;
}

// mean over tokens of [B*NTOK, C]
__global__ void pool_tok_kernel(const float* __restrict__ X, float* __restrict__ Y,
                                int C)
{
    int i = blockIdx.x * blockDim.x + threadIdx.x;
    if (i >= BSZ * C) return;
    const int b = i / C, c = i % C;
    float s = 0.f;
    for (int n = 0; n < NTOK; ++n) s += X[((size_t)b * NTOK + n) * C + c];
    Y[i] = s / (float)NTOK;
}

// copy packed [rows, Cs] into dst [rows, ldd] at column offset
__global__ void copy_cols_kernel(const float* __restrict__ S, int Cs,
                                 float* __restrict__ D, int ldd, int off,
                                 size_t total)
{
    size_t i = (size_t)blockIdx.x * blockDim.x + threadIdx.x;
    if (i >= total) return;
    const int c = (int)(i % Cs);
    const size_t row = i / Cs;
    D[row * ldd + off + c] = S[i];
}

__global__ void bn_relu_kernel(float* __restrict__ F, const float* __restrict__ g,
                               const float* __restrict__ b,
                               const float* __restrict__ m,
                               const float* __restrict__ v, int C, size_t total)
{
    size_t i = (size_t)blockIdx.x * blockDim.x + threadIdx.x;
    if (i >= total) return;
    const int c = (int)(i % C);
    float x = F[i];
    x = (x - m[c]) * rsqrtf(v[c] + 1e-5f) * g[c] + b[c];
    F[i] = fmaxf(x, 0.f);
}

// ---------------------------------------------------------------------------
// Host-side parameter structs (walk d_in in dict-insertion order)
// ---------------------------------------------------------------------------
struct MambaP  { const float *in_w,*in_b,*conv_w,*conv_b,*bc_w,*bc_b,*log_decay,*out_w,*out_b; };
struct MlpP    { const float *w1,*b1,*w2,*b2; };
struct MBlockP { const float *ln1_g,*ln1_b; MambaP mamba; const float *ln2_g,*ln2_b; MlpP mlp; };
struct CrossP  { const float *lnq_g,*lnq_b,*lnkv_g,*lnkv_b,*qw,*kw,*vw,*in_b,*ow,*ob,*ln2_g,*ln2_b; MlpP mlp; };
struct StageP  { MBlockP sa, sb; CrossP cab, cba; const float *alpha_ab,*alpha_ba; };

extern "C" void kernel_launch(void* const* d_in, const int* in_sizes, int n_in,
                              void* d_out, int out_size, void* d_ws, size_t ws_size,
                              hipStream_t stream)
{
    (void)in_sizes; (void)n_in; (void)out_size; (void)ws_size;
    const int Mtok = BSZ * NTOK;     // 43264 (multiple of 32)

    // ---- walk inputs / params -----------------------------------------
    int idx = 0;
    auto P = [&]() { return (const float*)d_in[idx++]; };
    const float* x1 = P();            // [256,144,13,13]
    const float* x2 = P();            // [256,4,13,13]
    const float *g1w=P(),*g1b=P(),*g2w=P(),*g2b=P();
    const float *ea_w=P(),*ea_b=P(),*eb_w=P(),*eb_b=P();
    const float *pos_a1=P(),*pos_b1=P(),*pos_a2=P(),*pos_b2=P(),*pos_a3=P(),*pos_b3=P();
    auto rdMamba = [&]() { MambaP m; m.in_w=P(); m.in_b=P(); m.conv_w=P(); m.conv_b=P();
                           m.bc_w=P(); m.bc_b=P(); m.log_decay=P(); m.out_w=P(); m.out_b=P(); return m; };
    auto rdMlp   = [&]() { MlpP m; m.w1=P(); m.b1=P(); m.w2=P(); m.b2=P(); return m; };
    auto rdMBlk  = [&]() { MBlockP m; m.ln1_g=P(); m.ln1_b=P(); m.mamba=rdMamba();
                           m.ln2_g=P(); m.ln2_b=P(); m.mlp=rdMlp(); return m; };
    auto rdCross = [&]() { CrossP c; c.lnq_g=P(); c.lnq_b=P(); c.lnkv_g=P(); c.lnkv_b=P();
                           c.qw=P(); c.kw=P(); c.vw=P(); c.in_b=P(); c.ow=P(); c.ob=P();
                           c.ln2_g=P(); c.ln2_b=P(); c.mlp=rdMlp(); return c; };
    auto rdStage = [&]() { StageP s; s.sa=rdMBlk(); s.sb=rdMBlk(); s.cab=rdCross();
                           s.cba=rdCross(); s.alpha_ab=P(); s.alpha_ba=P(); return s; };
    StageP st1 = rdStage(), st2 = rdStage(), st3 = rdStage();
    const float *pa12_w=P(),*pa12_b=P(),*pb12_w=P(),*pb12_b=P();
    const float *pa23_w=P(),*pa23_b=P(),*pb23_w=P(),*pb23_b=P();
    const float *fus_w=P(),*fus_b=P();
    const float *bn_g=P(),*bn_b=P(),*bn_m=P(),*bn_v=P();
    const float *fc_w=P(),*fc_b=P();

    // ---- workspace carving --------------------------------------------
    float* w = (float*)d_ws;
    size_t off = 0;
    auto carve = [&](size_t n) { float* p = w + off; off += n; return p; };
    float* Abuf  = carve((size_t)Mtok * 128);
    float* Abuf2 = carve((size_t)Mtok * 128);
    float* Bbuf  = carve((size_t)Mtok * 32);
    float* Bbuf2 = carve((size_t)Mtok * 32);
    float* L1    = carve((size_t)Mtok * 128);
    float* L2    = carve((size_t)Mtok * 128);
    float* T0    = carve((size_t)Mtok * 512);
    float* T1    = carve((size_t)Mtok * 512);
    float* T2    = carve((size_t)Mtok * 256);
    float* SM    = carve(65536);

    // ---- launch helpers ------------------------------------------------
    auto gemm = [&](const float* X, int ldx, const float* W, const float* bias,
                    const float* res, int ldres, const float* pos,
                    float* Y, int ldy, int Mt, int Kt, int Ft, int act) {
        dim3 grid((Ft + 31) / 32, (Mt + 31) / 32);
        gemm_wmma_kernel<<<grid, 32, 0, stream>>>(X, ldx, W, bias, res, ldres,
                                                  pos, Y, ldy, Mt, Kt, Ft, act);
    };
    auto ln = [&](const float* X, int D, const float* g, const float* b, float* Y) {
        ln_kernel<<<Mtok, 64, 0, stream>>>(X, g, b, Y, D);
    };
    auto ngrid = [](size_t n) { return dim3((unsigned)((n + 255) / 256)); };

    auto runMBlock = [&](float* Xb, int d, const MBlockP& p) {
        const int di = 2 * d;
        ln(Xb, d, p.ln1_g, p.ln1_b, L1);
        gemm(L1, d, p.mamba.in_w, p.mamba.in_b, nullptr, 0, nullptr,
             T0, 2 * di, Mtok, d, 2 * di, ACT_NONE);                  // [u|gate]
        const size_t tot = (size_t)Mtok * di;
        conv_silu_kernel<<<ngrid(tot), 256, 0, stream>>>(T0, 2 * di,
            p.mamba.conv_w, p.mamba.conv_b, T2, di, tot);             // uc
        gemm(T2, di, p.mamba.bc_w, p.mamba.bc_b, nullptr, 0, nullptr,
             T1, 2 * di, Mtok, di, 2 * di, ACT_NONE);                 // [bb|cc]
        mamba_scan_kernel<<<ngrid((size_t)BSZ * di), 256, 0, stream>>>(
            T1, 2 * di, T0 + di, 2 * di, p.mamba.log_decay, T2, di);  // y
        gemm(T2, di, p.mamba.out_w, p.mamba.out_b, Xb, d, nullptr,
             Xb, d, Mtok, di, d, ACT_NONE);                           // x += out
        ln(Xb, d, p.ln2_g, p.ln2_b, L1);
        gemm(L1, d, p.mlp.w1, p.mlp.b1, nullptr, 0, nullptr,
             T0, 4 * d, Mtok, d, 4 * d, ACT_GELU);
        gemm(T0, 4 * d, p.mlp.w2, p.mlp.b2, Xb, d, nullptr,
             Xb, d, Mtok, 4 * d, d, ACT_NONE);                        // x += mlp
    };

    auto runCross = [&](float* Xt, int E, const float* Xs, int S,
                        const CrossP& p, const float* alpha) {
        const int dh = E / 8;
        ln(Xt, E, p.lnq_g, p.lnq_b, L1);
        ln(Xs, S, p.lnkv_g, p.lnkv_b, L2);
        float* Qb = T0;
        float* Kb = T0 + (size_t)Mtok * 128;
        float* Vb = T0 + (size_t)Mtok * 256;
        gemm(L1, E, p.qw, p.in_b,         nullptr, 0, nullptr, Qb, E, Mtok, E, E, ACT_NONE);
        gemm(L2, S, p.kw, p.in_b + E,     nullptr, 0, nullptr, Kb, E, Mtok, S, E, ACT_NONE);
        gemm(L2, S, p.vw, p.in_b + 2 * E, nullptr, 0, nullptr, Vb, E, Mtok, S, E, ACT_NONE);
        mha_kernel<<<BSZ * 8, 192, 0, stream>>>(Qb, Kb, Vb, T2, E, dh);
        gemm(T2, E, p.ow, p.ob, nullptr, 0, nullptr, T1, E, Mtok, E, E, ACT_NONE); // o
        const size_t tot = (size_t)Mtok * E;
        add_kernel<<<ngrid(tot), 256, 0, stream>>>(T2, Xt, T1, tot);  // xt2 = xt + o
        ln(T2, E, p.ln2_g, p.ln2_b, L2);
        gemm(L2, E, p.mlp.w1, p.mlp.b1, nullptr, 0, nullptr,
             T0, 4 * E, Mtok, E, 4 * E, ACT_GELU);
        gemm(T0, 4 * E, p.mlp.w2, p.mlp.b2, nullptr, 0, nullptr,
             T2, E, Mtok, 4 * E, E, ACT_NONE);                        // m
        blend_kernel<<<ngrid(tot), 256, 0, stream>>>(Xt, T1, T2, alpha, tot);
    };

    auto runStage = [&](float* Ab, int da, float* Bb, int db, const StageP& s) {
        runMBlock(Ab, da, s.sa);
        runMBlock(Bb, db, s.sb);
        runCross(Ab, da, Bb, db, s.cab, s.alpha_ab);
        runCross(Bb, db, Ab, da, s.cba, s.alpha_ba);
    };

    // ---- 0. HSI band gating -------------------------------------------
    float* pooled = SM;                // [256,4]
    float* hgate  = SM + 4096;         // [256,16]
    float* ggate  = SM + 16384;        // [256,144]
    pool_hw_kernel<<<ngrid(BSZ * 4), 256, 0, stream>>>(x2, pooled, 4);
    gemm(pooled, 4,  g1w, g1b, nullptr, 0, nullptr, hgate, 16,  BSZ, 4,  16,  ACT_RELU);
    gemm(hgate,  16, g2w, g2b, nullptr, 0, nullptr, ggate, 144, BSZ, 16, 144, ACT_SIGMOID);

    // ---- 1. tokenization ----------------------------------------------
    tok_transpose_kernel<<<ngrid((size_t)Mtok * 144), 256, 0, stream>>>(
        x1, ggate, T1, 144, (size_t)Mtok * 144);
    gemm(T1, 144, ea_w, ea_b, nullptr, 0, pos_a1, Abuf, 128, Mtok, 144, 128, ACT_NONE);
    tok_transpose_kernel<<<ngrid((size_t)Mtok * 4), 256, 0, stream>>>(
        x2, nullptr, T1, 4, (size_t)Mtok * 4);
    gemm(T1, 4, eb_w, eb_b, nullptr, 0, pos_b1, Bbuf, 8, Mtok, 4, 8, ACT_NONE);

    // ---- 2. stages with inter-stage projections ------------------------
    runStage(Abuf, 128, Bbuf, 8, st1);
    gemm(Abuf, 128, pa12_w, pa12_b, nullptr, 0, pos_a2, Abuf2, 64, Mtok, 128, 64, ACT_NONE);
    gemm(Bbuf, 8,   pb12_w, pb12_b, nullptr, 0, pos_b2, Bbuf2, 16, Mtok, 8,   16, ACT_NONE);
    std::swap(Abuf, Abuf2); std::swap(Bbuf, Bbuf2);

    runStage(Abuf, 64, Bbuf, 16, st2);
    gemm(Abuf, 64, pa23_w, pa23_b, nullptr, 0, pos_a3, Abuf2, 32, Mtok, 64, 32, ACT_NONE);
    gemm(Bbuf, 16, pb23_w, pb23_b, nullptr, 0, pos_b3, Bbuf2, 32, Mtok, 16, 32, ACT_NONE);
    std::swap(Abuf, Abuf2); std::swap(Bbuf, Bbuf2);

    runStage(Abuf, 32, Bbuf, 32, st3);

    // ---- 3. fuse -> BN/ReLU -> pool -> FC ------------------------------
    copy_cols_kernel<<<ngrid((size_t)Mtok * 32), 256, 0, stream>>>(
        Abuf, 32, T0, 64, 0,  (size_t)Mtok * 32);
    copy_cols_kernel<<<ngrid((size_t)Mtok * 32), 256, 0, stream>>>(
        Bbuf, 32, T0, 64, 32, (size_t)Mtok * 32);
    gemm(T0, 64, fus_w, fus_b, nullptr, 0, nullptr, T1, 32, Mtok, 64, 32, ACT_NONE);
    bn_relu_kernel<<<ngrid((size_t)Mtok * 32), 256, 0, stream>>>(
        T1, bn_g, bn_b, bn_m, bn_v, 32, (size_t)Mtok * 32);
    pool_tok_kernel<<<ngrid(BSZ * 32), 256, 0, stream>>>(T1, SM, 32);
    gemm(SM, 32, fc_w, fc_b, nullptr, 0, nullptr,
         (float*)d_out, 15, BSZ, 32, 15, ACT_NONE);   // [256, 15] logits
}
